// cascade_sg_first_contextGAT_Decoder_11166914970420
// MI455X (gfx1250) — matrix operand order repaired and checked by reading, not compiled, not run on hardware
//
// CDNA5 (gfx1250) cascade contextGAT decoder forward.
//
// - All weights converted to bf16 once per launch (~137MB -> L2-resident on
//   the 192MB L2 across the 16 scan steps); GEMMs use v_wmma_f32_16x16x32_bf16
//   with fp32 accumulation (memory-bound workload: bf16 halves weight bytes).
// - Vocab weight matrices zero-padded to N=9536 so the WMMA inner loop is
//   branch-free (no per-element edge guards in the hot loop).
// - NT=4 N-tiles per wave: one A fragment feeds 4 WMMAs (4x less A re-read).
// - Skinny M=32 GEMMs use a deterministic K-split (S=4) into fp32 partials +
//   a combine kernel (sum partials + biases); the two LSTM GEMMs share one
//   partial buffer and one combine.
// - Hoisted: image-attention feature GEMM (time-invariant); skipped: Wphi_e
//   edge update (dead with k_steps=1).

#include <hip/hip_runtime.h>

typedef __bf16 bf16;
typedef __attribute__((ext_vector_type(16))) __bf16 v16bf;
typedef __attribute__((ext_vector_type(8)))  float  v8f;

// ---- problem dimensions ----
constexpr int kB = 32, kR = 36, kNOBJ = 36, kNREL = 64;
constexpr int kFD = 2048, kGD = 512, kDD = 1024, kEMB = 1024, kAD = 1024;
constexpr int kV = 9490, kT = 16;
constexpr int kX1 = kDD + kFD + kGD + kEMB;   // 4608
constexpr int kX2 = kGD + kFD + kDD;          // 3584
constexpr int kBN = kB * kNOBJ;               // 1152
constexpr int k4DD = 4 * kDD;                 // 4096
constexpr int kNT = 4;                        // N-tiles per wave
constexpr int kSPLIT = 4;                     // K-split for skinny GEMMs
constexpr int kVpad = ((kV + kNT * 16 - 1) / (kNT * 16)) * (kNT * 16);  // 9536

__device__ __forceinline__ bf16 to_bf16(float x) { return (bf16)x; }
__device__ __forceinline__ float sigf(float x) { return 1.f / (1.f + expf(-x)); }

// A fragment: 16-bit A-matrix 16x32 per-lane layout (ISA 7.12.2), fp32->bf16
__device__ __forceinline__ v16bf load_a_frag(const float* __restrict__ Arow,
                                             int k0, int half) {
  v16bf a;
#pragma unroll
  for (int r = 0; r < 8; ++r) {
    const int kb = (r < 4) ? (2 * r + 8 * half) : (16 + 2 * (r - 4) + 8 * half);
    a[2 * r]     = to_bf16(Arow[k0 + kb]);
    a[2 * r + 1] = to_bf16(Arow[k0 + kb + 1]);
  }
  return a;
}

// ===================== fused WMMA GEMM (large-M GEMMs) =====================
// D[M,N] = act( A[M,K](f32) @ B[K,N](bf16, width ldb >= padded N) + bias )
// K % 32 == 0; B readable (zero-padded) up to ceil(tilesN/NT)*NT*16 columns.
template <bool RELU, int NT>
__global__ void wmma_gemm_kernel(const float* __restrict__ A,
                                 const bf16* __restrict__ B,
                                 const float* __restrict__ bias,
                                 float* __restrict__ D,
                                 int M, int N, int K, int lda, int ldb, int ldd) {
  const int tilesM = (M + 15) >> 4;
  const int tilesN = (N + 15) >> 4;
  const int ntG = (tilesN + NT - 1) / NT;
  const int id = blockIdx.x * (blockDim.x >> 5) + (threadIdx.x >> 5);
  if (id >= tilesM * ntG) return;                 // wave-uniform
  const int tm = id / ntG, tg = id % ntG;
  const int row0 = tm << 4, n0 = tg * (NT * 16);
  const int lane = threadIdx.x & 31;
  const int m = lane & 15, half = lane >> 4;

  v8f zero = {};
  v8f acc[NT];
#pragma unroll
  for (int u = 0; u < NT; ++u) acc[u] = zero;

  const float* __restrict__ Arow = A + (size_t)(row0 + m) * lda;
  for (int k0 = 0; k0 < K; k0 += 32) {
    v16bf a = load_a_frag(Arow, k0, half);
    const bf16* __restrict__ Brow = B + (size_t)k0 * ldb + n0 + (lane & 15);
#pragma unroll
    for (int u = 0; u < NT; ++u) {
      v16bf b;
#pragma unroll
      for (int j = 0; j < 16; ++j) b[j] = Brow[(size_t)((j < 8) ? (8 * half + j) : (16 + 8 * half + (j - 8))) * ldb + u * 16];
      acc[u] = __builtin_amdgcn_wmma_f32_16x16x32_bf16(false, a, false, b,
                                                       (short)0, acc[u], false, false);
    }
  }
#pragma unroll
  for (int u = 0; u < NT; ++u) {
    const int col = n0 + u * 16 + (lane & 15);
    if (col < N) {
      const float bv = bias ? bias[col] : 0.f;
#pragma unroll
      for (int r = 0; r < 8; ++r) {
        const int row = row0 + r + 8 * half;
        if (row < M) {
          float v = acc[u][r] + bv;
          if (RELU) v = fmaxf(v, 0.f);
          D[(size_t)row * ldd + col] = v;
        }
      }
    }
  }
}

// ============== K-split WMMA GEMM (skinny M=32 GEMMs) ==============
// Partial s: P[s*M .. ] = A[:, s*K/S:(s+1)*K/S] @ B[...]. ldp == ldb (padded).
template <int NT>
__global__ void wmma_gemm_split_kernel(const float* __restrict__ A,
                                       const bf16* __restrict__ B,
                                       float* __restrict__ P,
                                       int M, int K, int lda, int ldb,
                                       int S, int tilesM, int ntG) {
  const int id = blockIdx.x * (blockDim.x >> 5) + (threadIdx.x >> 5);
  if (id >= tilesM * ntG * S) return;             // wave-uniform
  const int grp = id % (tilesM * ntG);
  const int sidx = id / (tilesM * ntG);
  const int tm = grp / ntG, tg = grp % ntG;
  const int row0 = tm << 4, n0 = tg * (NT * 16);
  const int lane = threadIdx.x & 31;
  const int m = lane & 15, half = lane >> 4;

  v8f zero = {};
  v8f acc[NT];
#pragma unroll
  for (int u = 0; u < NT; ++u) acc[u] = zero;

  const int kc = K / S;
  const int kend = (sidx + 1) * kc;
  const float* __restrict__ Arow = A + (size_t)(row0 + m) * lda;
  for (int k0 = sidx * kc; k0 < kend; k0 += 32) {
    v16bf a = load_a_frag(Arow, k0, half);
    const bf16* __restrict__ Brow = B + (size_t)k0 * ldb + n0 + (lane & 15);
#pragma unroll
    for (int u = 0; u < NT; ++u) {
      v16bf b;
#pragma unroll
      for (int j = 0; j < 16; ++j) b[j] = Brow[(size_t)((j < 8) ? (8 * half + j) : (16 + 8 * half + (j - 8))) * ldb + u * 16];
      acc[u] = __builtin_amdgcn_wmma_f32_16x16x32_bf16(false, a, false, b,
                                                       (short)0, acc[u], false, false);
    }
  }
  float* __restrict__ Pr = P + (size_t)sidx * M * ldb;
#pragma unroll
  for (int u = 0; u < NT; ++u) {
    const int col = n0 + u * 16 + (lane & 15);
#pragma unroll
    for (int r = 0; r < 8; ++r) {
      const int row = row0 + r + 8 * half;   // M is a multiple of 16 here
      Pr[(size_t)row * ldb + col] = acc[u][r];
    }
  }
}

// D[m,n] = sum_s P[s,m,n] + b1[n] + b2[n]
__global__ void combine_kernel(const float* __restrict__ P, int S, int M, int N, int ldp,
                               const float* __restrict__ b1, const float* __restrict__ b2,
                               float* __restrict__ D, int ldd) {
  int i = blockIdx.x * blockDim.x + threadIdx.x;
  if (i >= M * N) return;
  int mrow = i / N, n = i % N;
  float s = 0.f;
  if (b1) s += b1[n];
  if (b2) s += b2[n];
  for (int ss = 0; ss < S; ++ss) s += P[((size_t)ss * M + mrow) * ldp + n];
  D[(size_t)mrow * ldd + n] = s;
}

// ===================== conversion / elementwise kernels =====================
__global__ void convert_kernel(const float* __restrict__ s, bf16* __restrict__ d, size_t n) {
  size_t i = (size_t)blockIdx.x * blockDim.x + threadIdx.x;
  if (i < n) d[i] = to_bf16(s[i]);
}
// [rows,N] f32 -> [rows,ldp] bf16, zero-padded columns N..ldp
__global__ void convert_pad_kernel(const float* __restrict__ s, bf16* __restrict__ d,
                                   int rows, int N, int ldp) {
  size_t i = (size_t)blockIdx.x * blockDim.x + threadIdx.x;
  if (i >= (size_t)rows * ldp) return;
  int r = (int)(i / ldp), n = (int)(i % ldp);
  d[i] = (n < N) ? to_bf16(s[(size_t)r * N + n]) : to_bf16(0.f);
}
// src [rows,cols] -> dst [cols,rows] bf16 (materialize W^T)
__global__ void convertT_kernel(const float* __restrict__ s, bf16* __restrict__ d,
                                int rows, int cols) {
  size_t i = (size_t)blockIdx.x * blockDim.x + threadIdx.x;
  if (i >= (size_t)rows * cols) return;
  int r = (int)(i / cols), c = (int)(i % cols);
  d[(size_t)c * rows + r] = to_bf16(s[i]);
}
__global__ void mean_kernel(const float* __restrict__ src, float* __restrict__ dst,
                            int Nf, int Df) {
  int i = blockIdx.x * blockDim.x + threadIdx.x;
  if (i >= kB * Df) return;
  int b = i / Df, d = i % Df;
  float s = 0.f;
  for (int r = 0; r < Nf; ++r) s += src[((size_t)b * Nf + r) * Df + d];
  dst[i] = s / (float)Nf;
}
__global__ void build_x1_kernel(const float* __restrict__ h2, const float* __restrict__ imgm,
                                const float* __restrict__ objm, const float* __restrict__ emb,
                                const int* __restrict__ caps, int t, float* __restrict__ x1) {
  int i = blockIdx.x * blockDim.x + threadIdx.x;
  if (i >= kB * kX1) return;
  int b = i / kX1, d = i % kX1;
  float v;
  if (d < kDD) v = h2[b * kDD + d];
  else if (d < kDD + kFD) v = imgm[b * kFD + (d - kDD)];
  else if (d < kDD + kFD + kGD) v = objm[b * kGD + (d - kDD - kFD)];
  else {
    int tok = caps[b * kT + t];
    v = emb[(size_t)tok * kEMB + (d - kDD - kFD - kGD)];
  }
  x1[i] = v;
}
__global__ void build_x2_kernel(const float* __restrict__ g, const float* __restrict__ v,
                                const float* __restrict__ h1, float* __restrict__ x2) {
  int i = blockIdx.x * blockDim.x + threadIdx.x;
  if (i >= kB * kX2) return;
  int b = i / kX2, d = i % kX2;
  float val;
  if (d < kGD) val = g[b * kGD + d];
  else if (d < kGD + kFD) val = v[b * kFD + (d - kGD)];
  else val = h1[b * kDD + (d - kGD - kFD)];
  x2[i] = val;
}
__global__ void lstm_pointwise_kernel(const float* __restrict__ g,
                                      float* __restrict__ h, float* __restrict__ c) {
  int i = blockIdx.x * blockDim.x + threadIdx.x;
  if (i >= kB * kDD) return;
  int b = i / kDD, d = i % kDD;
  const float* gb = g + (size_t)b * k4DD;
  float ig = sigf(gb[d]), fg = sigf(gb[kDD + d]);
  float gg = tanhf(gb[2 * kDD + d]), og = sigf(gb[3 * kDD + d]);
  float c2 = fg * c[i] + ig * gg;
  c[i] = c2;
  h[i] = og * tanhf(c2);
}
// s[b,i] = dot(h_t[b], w[:GD]) + dot(feats[b,i], w[GD:])
__global__ void score_kernel(const float* __restrict__ ht, const float* __restrict__ feats,
                             const float* __restrict__ w, int Nf, float* __restrict__ out) {
  int i = blockIdx.x * blockDim.x + threadIdx.x;
  if (i >= kB * Nf) return;
  int b = i / Nf;
  const float* f = feats + (size_t)i * kGD;
  const float* h = ht + (size_t)b * kGD;
  float s = 0.f;
  for (int d = 0; d < kGD; ++d) s += h[d] * w[d] + f[d] * w[kGD + d];
  out[i] = s;
}
// segment softmax aggregation: one block per (b, node), 128 threads = 2*NREL msgs
__global__ void gat_aggregate_kernel(const float* __restrict__ s_n, const float* __restrict__ s_e,
                                     const float* __restrict__ Fn, const float* __restrict__ Fe,
                                     const int* __restrict__ pair_ids,
                                     const unsigned char* __restrict__ rmask,
                                     float* __restrict__ applied, int* __restrict__ cnt_out) {
  int bn = blockIdx.x;
  int b = bn / kNOBJ, n = bn % kNOBJ;
  __shared__ float w_sh[2 * kNREL];
  __shared__ float red[128];
  __shared__ int src_sh[kNREL];
  int tid = threadIdx.x;
  int e = tid & (kNREL - 1);
  bool isEdge = tid >= kNREL;
  int pbase = (b * kNREL + e) * 2;
  int srcid = pair_ids[pbase + 0];
  int dstid = pair_ids[pbase + 1];
  if (tid < kNREL) src_sh[tid] = srcid;
  bool delivered = (dstid == n);
  bool msk = rmask[b * kNREL + e] != 0;
  bool valid = delivered && msk;
  float sraw = isEdge ? s_e[b * kNREL + e] : s_n[b * kNOBJ + srcid];
  float sc = delivered ? (msk ? sraw : -1e9f) : -INFINITY;
  red[tid] = sc; __syncthreads();
  for (int st = 64; st > 0; st >>= 1) { if (tid < st) red[tid] = fmaxf(red[tid], red[tid + st]); __syncthreads(); }
  float mx = red[0]; __syncthreads();
  if (!(mx > -INFINITY)) mx = 0.f;   // empty segment -> 0 (jnp.isfinite fixup)
  float w = valid ? expf(sc - mx) : 0.f;
  w_sh[tid] = w;
  red[tid] = w; __syncthreads();
  for (int st = 64; st > 0; st >>= 1) { if (tid < st) red[tid] += red[tid + st]; __syncthreads(); }
  float den = red[0]; __syncthreads();
  red[tid] = valid ? 1.f : 0.f; __syncthreads();
  for (int st = 64; st > 0; st >>= 1) { if (tid < st) red[tid] += red[tid + st]; __syncthreads(); }
  float cnt = red[0]; __syncthreads();
  float inv = 1.f / fmaxf(den, 1e-9f);
  for (int d = tid; d < kGD; d += 128) {
    float acc = 0.f;
    for (int ee = 0; ee < kNREL; ++ee) {
      float w1 = w_sh[ee];
      float w2 = w_sh[kNREL + ee];
      if (w1 != 0.f) acc += w1 * Fn[((size_t)b * kNOBJ + src_sh[ee]) * kGD + d];
      if (w2 != 0.f) acc += w2 * Fe[((size_t)b * kNREL + ee) * kGD + d];
    }
    applied[(size_t)bn * kGD + d] = acc * inv;
  }
  if (tid == 0) cnt_out[bn] = (cnt > 0.f) ? 1 : 0;
}
__global__ void concat2_kernel(const float* __restrict__ A, const float* __restrict__ B,
                               float* __restrict__ out, int rows, int da, int db) {
  int dt = da + db;
  size_t i = (size_t)blockIdx.x * blockDim.x + threadIdx.x;
  if (i >= (size_t)rows * dt) return;
  int r = (int)(i / dt), d = (int)(i % dt);
  out[i] = (d < da) ? A[(size_t)r * da + d] : B[(size_t)r * db + (d - da)];
}
__global__ void mask_rows_kernel(float* __restrict__ io, const int* __restrict__ cnt,
                                 int rows, int cols) {
  size_t i = (size_t)blockIdx.x * blockDim.x + threadIdx.x;
  if (i >= (size_t)rows * cols) return;
  if (cnt[i / cols] == 0) io[i] = 0.f;
}
__global__ void row_sum_mask_kernel(const float* __restrict__ io, unsigned char* __restrict__ m,
                                    int rows, int cols) {
  int r = blockIdx.x * blockDim.x + threadIdx.x;
  if (r >= rows) return;
  float s = 0.f;
  for (int d = 0; d < cols; ++d) s += io[(size_t)r * cols + d];
  m[r] = (s != 0.f) ? 1 : 0;
}
// sc[b,i] = dot(relu(ff[b,i,:]+hd[b,:]), wfull) + bfull ; masked -> -1e9
__global__ void att_score_kernel(const float* __restrict__ ff, const float* __restrict__ hd,
                                 const float* __restrict__ wfull, const float* __restrict__ bfull,
                                 const unsigned char* __restrict__ mask,
                                 float* __restrict__ sc, int Nf, int Dh) {
  int i = blockIdx.x * blockDim.x + threadIdx.x;
  if (i >= kB * Nf) return;
  int b = i / Nf;
  const float* f = ff + (size_t)i * Dh;
  const float* h = hd + (size_t)b * Dh;
  float s = 0.f;
  for (int d = 0; d < Dh; ++d) s += fmaxf(f[d] + h[d], 0.f) * wfull[d];
  s += bfull[0];
  if (mask && !mask[i]) s = -1e9f;
  sc[i] = s;
}
__global__ void softmax_rows_kernel(float* __restrict__ sc, int rows, int Nf) {
  int b = blockIdx.x * blockDim.x + threadIdx.x;
  if (b >= rows) return;
  float* s = sc + (size_t)b * Nf;
  float mx = -INFINITY;
  for (int i = 0; i < Nf; ++i) mx = fmaxf(mx, s[i]);
  float den = 0.f;
  for (int i = 0; i < Nf; ++i) { float e = expf(s[i] - mx); s[i] = e; den += e; }
  float inv = 1.f / den;
  for (int i = 0; i < Nf; ++i) s[i] *= inv;
}
__global__ void weighted_sum_kernel(const float* __restrict__ feats, const float* __restrict__ alpha,
                                    float* __restrict__ out, int Nf, int Df) {
  int i = blockIdx.x * blockDim.x + threadIdx.x;
  if (i >= kB * Df) return;
  int b = i / Df, d = i % Df;
  float s = 0.f;
  for (int r = 0; r < Nf; ++r) s += feats[((size_t)b * Nf + r) * Df + d] * alpha[b * Nf + r];
  out[i] = s;
}

// ===================== host side =====================
static inline int cdiv(long long a, long long b) { return (int)((a + b - 1) / b); }

static void gemm_fused(const float* A, const bf16* B, const float* bias, float* D,
                       int M, int N, int K, int lda, int ldb, int ldd,
                       bool relu, hipStream_t s) {
  int waves = cdiv(M, 16) * cdiv(cdiv(N, 16), kNT);
  dim3 blk(128), grd(cdiv(waves, 4));
  if (relu)
    wmma_gemm_kernel<true, kNT><<<grd, blk, 0, s>>>(A, B, bias, D, M, N, K, lda, ldb, ldd);
  else
    wmma_gemm_kernel<false, kNT><<<grd, blk, 0, s>>>(A, B, bias, D, M, N, K, lda, ldb, ldd);
}
// one K-split partial GEMM (S=kSPLIT partials written at P)
static void gemm_split(const float* A, const bf16* B, float* P,
                       int M, int N, int K, int lda, int ldb, hipStream_t s) {
  int tilesM = cdiv(M, 16), ntG = cdiv(cdiv(N, 16), kNT);
  int waves = tilesM * ntG * kSPLIT;
  dim3 blk(128), grd(cdiv(waves, 4));
  wmma_gemm_split_kernel<kNT><<<grd, blk, 0, s>>>(A, B, P, M, K, lda, ldb,
                                                  kSPLIT, tilesM, ntG);
}
static void combine(const float* P, int S, int M, int N, int ldp,
                    const float* b1, const float* b2, float* D, int ldd, hipStream_t s) {
  dim3 blk(256), grd(cdiv((long long)M * N, 256));
  combine_kernel<<<grd, blk, 0, s>>>(P, S, M, N, ldp, b1, b2, D, ldd);
}

extern "C" void kernel_launch(void* const* d_in, const int* in_sizes, int n_in,
                              void* d_out, int out_size, void* d_ws, size_t ws_size,
                              hipStream_t stream) {
  // ---- inputs ----
  const float* image_features   = (const float*)d_in[0];
  const float* object_features  = (const float*)d_in[1];
  const float* relation_features= (const float*)d_in[2];
  const float* emb     = (const float*)d_in[3];
  const float* Wih1    = (const float*)d_in[4];
  const float* Whh1    = (const float*)d_in[5];
  const float* bih1    = (const float*)d_in[6];
  const float* bhh1    = (const float*)d_in[7];
  const float* Wih2    = (const float*)d_in[8];
  const float* Whh2    = (const float*)d_in[9];
  const float* bih2    = (const float*)d_in[10];
  const float* bhh2    = (const float*)d_in[11];
  const float* a1_Wf   = (const float*)d_in[12];
  const float* a1_bf   = (const float*)d_in[13];
  const float* a1_Wd   = (const float*)d_in[14];
  const float* a1_bd   = (const float*)d_in[15];
  const float* a1_wfull= (const float*)d_in[16];
  const float* a1_bfull= (const float*)d_in[17];
  const float* a2_Wf   = (const float*)d_in[18];
  const float* a2_bf   = (const float*)d_in[19];
  const float* a2_Wd   = (const float*)d_in[20];
  const float* a2_bd   = (const float*)d_in[21];
  const float* a2_wfull= (const float*)d_in[22];
  const float* a2_bfull= (const float*)d_in[23];
  const float* cg_Wproj= (const float*)d_in[24];
  const float* cg_wobj = (const float*)d_in[25];
  const float* cg_wrel = (const float*)d_in[26];
  /* d_in[27] cg_Wphi_e: dead with k_steps=1 */
  const float* cg_Wphi_n=(const float*)d_in[28];
  const float* Wfc     = (const float*)d_in[29];
  const float* bfc     = (const float*)d_in[30];
  const float* Wfc1    = (const float*)d_in[31];
  const float* bfc1    = (const float*)d_in[32];
  const int*   pair_ids= (const int*)d_in[33];
  const int*   caps    = (const int*)d_in[34];
  const unsigned char* rel_mask = (const unsigned char*)d_in[36];
  float* out = (float*)d_out;
  (void)ws_size; (void)n_in; (void)in_sizes; (void)out_size;

  // ---- workspace bump allocator ----
  char* ws = (char*)d_ws;
  size_t off = 0;
  auto alloc = [&](size_t bytes) -> void* {
    void* p = ws + off;
    off = (off + bytes + 255) & ~(size_t)255;
    return p;
  };
  // bf16 weights (transposed where reference does x @ W.T; vocab padded)
  bf16* wih1t = (bf16*)alloc((size_t)kX1 * k4DD * 2);
  bf16* whh1t = (bf16*)alloc((size_t)kDD * k4DD * 2);
  bf16* wih2t = (bf16*)alloc((size_t)kX2 * k4DD * 2);
  bf16* whh2t = (bf16*)alloc((size_t)kDD * k4DD * 2);
  bf16* wfc_b  = (bf16*)alloc((size_t)kDD * kVpad * 2);
  bf16* wfc1_b = (bf16*)alloc((size_t)kDD * kVpad * 2);
  bf16* a1wf_b = (bf16*)alloc((size_t)kGD * kAD * 2);
  bf16* a1wd_b = (bf16*)alloc((size_t)kDD * kAD * 2);
  bf16* a2wf_b = (bf16*)alloc((size_t)kFD * kAD * 2);
  bf16* a2wd_b = (bf16*)alloc((size_t)(kDD + kGD) * kAD * 2);
  bf16* wproj_b= (bf16*)alloc((size_t)kDD * kGD * 2);
  bf16* wphin_b= (bf16*)alloc((size_t)2 * kGD * kGD * 2);
  // K-split partial buffer (max: 2*SPLIT partials of [32 x kVpad])
  float* part = (float*)alloc((size_t)2 * kSPLIT * kB * kVpad * 4);
  // f32 scratch
  float* imgm = (float*)alloc((size_t)kB * kFD * 4);
  float* objm = (float*)alloc((size_t)kB * kGD * 4);
  float* fA2  = (float*)alloc((size_t)kB * kR * kAD * 4);
  float* h1 = (float*)alloc((size_t)kB * kDD * 4);
  float* c1 = (float*)alloc((size_t)kB * kDD * 4);
  float* h2 = (float*)alloc((size_t)kB * kDD * 4);
  float* c2 = (float*)alloc((size_t)kB * kDD * 4);
  float* x1 = (float*)alloc((size_t)kB * kX1 * 4);
  float* g1 = (float*)alloc((size_t)kB * k4DD * 4);
  float* ht = (float*)alloc((size_t)kB * kGD * 4);
  float* s_n = (float*)alloc((size_t)kB * kNOBJ * 4);
  float* s_e = (float*)alloc((size_t)kB * kNREL * 4);
  float* applied = (float*)alloc((size_t)kBN * kGD * 4);
  int*   cnt = (int*)alloc((size_t)kBN * 4);
  float* catn = (float*)alloc((size_t)kBN * 2 * kGD * 4);
  float* io   = (float*)alloc((size_t)kBN * kGD * 4);
  unsigned char* io_mask = (unsigned char*)alloc(kBN);
  float* fa1 = (float*)alloc((size_t)kBN * kAD * 4);
  float* hd1 = (float*)alloc((size_t)kB * kAD * 4);
  float* sc1 = (float*)alloc((size_t)kB * kNOBJ * 4);
  float* gvec = (float*)alloc((size_t)kB * kGD * 4);
  float* cat_hg = (float*)alloc((size_t)kB * (kDD + kGD) * 4);
  float* hd2 = (float*)alloc((size_t)kB * kAD * 4);
  float* sc2 = (float*)alloc((size_t)kB * kR * 4);
  float* vvec = (float*)alloc((size_t)kB * kFD * 4);
  float* x2 = (float*)alloc((size_t)kB * kX2 * 4);
  float* g2 = (float*)alloc((size_t)kB * k4DD * 4);

  const int TPB = 256;
  auto elts = [&](size_t n) { return dim3(cdiv((long long)n, TPB)); };
  float* partB = part + (size_t)kSPLIT * kB * k4DD;   // second partial group (LSTM hh)

  // ---- 1) weight conversion (once per launch, amortized over 16 steps) ----
  convertT_kernel<<<elts((size_t)k4DD * kX1), TPB, 0, stream>>>(Wih1, wih1t, k4DD, kX1);
  convertT_kernel<<<elts((size_t)k4DD * kDD), TPB, 0, stream>>>(Whh1, whh1t, k4DD, kDD);
  convertT_kernel<<<elts((size_t)k4DD * kX2), TPB, 0, stream>>>(Wih2, wih2t, k4DD, kX2);
  convertT_kernel<<<elts((size_t)k4DD * kDD), TPB, 0, stream>>>(Whh2, whh2t, k4DD, kDD);
  convert_pad_kernel<<<elts((size_t)kDD * kVpad), TPB, 0, stream>>>(Wfc,  wfc_b,  kDD, kV, kVpad);
  convert_pad_kernel<<<elts((size_t)kDD * kVpad), TPB, 0, stream>>>(Wfc1, wfc1_b, kDD, kV, kVpad);
  convert_kernel<<<elts((size_t)kGD * kAD), TPB, 0, stream>>>(a1_Wf, a1wf_b, (size_t)kGD * kAD);
  convert_kernel<<<elts((size_t)kDD * kAD), TPB, 0, stream>>>(a1_Wd, a1wd_b, (size_t)kDD * kAD);
  convert_kernel<<<elts((size_t)kFD * kAD), TPB, 0, stream>>>(a2_Wf, a2wf_b, (size_t)kFD * kAD);
  convert_kernel<<<elts((size_t)(kDD + kGD) * kAD), TPB, 0, stream>>>(a2_Wd, a2wd_b, (size_t)(kDD + kGD) * kAD);
  convert_kernel<<<elts((size_t)kDD * kGD), TPB, 0, stream>>>(cg_Wproj, wproj_b, (size_t)kDD * kGD);
  convert_kernel<<<elts((size_t)2 * kGD * kGD), TPB, 0, stream>>>(cg_Wphi_n, wphin_b, (size_t)2 * kGD * kGD);

  // ---- 2) time-invariant precompute ----
  mean_kernel<<<elts(kB * kFD), TPB, 0, stream>>>(image_features, imgm, kR, kFD);
  mean_kernel<<<elts(kB * kGD), TPB, 0, stream>>>(object_features, objm, kNOBJ, kGD);
  gemm_fused(image_features, a2wf_b, a2_bf, fA2, kB * kR, kAD, kFD, kFD, kAD, kAD, false, stream);

  hipMemsetAsync(h1, 0, (size_t)kB * kDD * 4, stream);
  hipMemsetAsync(c1, 0, (size_t)kB * kDD * 4, stream);
  hipMemsetAsync(h2, 0, (size_t)kB * kDD * 4, stream);
  hipMemsetAsync(c2, 0, (size_t)kB * kDD * 4, stream);

  // ---- 3) T=16 sequential scan ----
  for (int t = 0; t < kT; ++t) {
    build_x1_kernel<<<elts(kB * kX1), TPB, 0, stream>>>(h2, imgm, objm, emb, caps, t, x1);
    // LSTM1: g1 = x1@Wih1.T + h1@Whh1.T + bih1 + bhh1 (2*S partials, 1 combine)
    gemm_split(x1, wih1t, part,  kB, k4DD, kX1, kX1, k4DD, stream);
    gemm_split(h1, whh1t, partB, kB, k4DD, kDD, kDD, k4DD, stream);
    combine(part, 2 * kSPLIT, kB, k4DD, k4DD, bih1, bhh1, g1, k4DD, stream);
    lstm_pointwise_kernel<<<elts(kB * kDD), TPB, 0, stream>>>(g1, h1, c1);
    // contextGAT
    gemm_split(h1, wproj_b, part, kB, kGD, kDD, kDD, kGD, stream);
    combine(part, kSPLIT, kB, kGD, kGD, nullptr, nullptr, ht, kGD, stream);
    score_kernel<<<elts(kB * kNOBJ), TPB, 0, stream>>>(ht, object_features, cg_wobj, kNOBJ, s_n);
    score_kernel<<<elts(kB * kNREL), TPB, 0, stream>>>(ht, relation_features, cg_wrel, kNREL, s_e);
    gat_aggregate_kernel<<<dim3(kBN), dim3(128), 0, stream>>>(
        s_n, s_e, object_features, relation_features, pair_ids, rel_mask, applied, cnt);
    concat2_kernel<<<elts((size_t)kBN * 2 * kGD), TPB, 0, stream>>>(
        applied, object_features, catn, kBN, kGD, kGD);
    gemm_fused(catn, wphin_b, nullptr, io, kBN, kGD, 2 * kGD, 2 * kGD, kGD, kGD, true, stream);
    mask_rows_kernel<<<elts((size_t)kBN * kGD), TPB, 0, stream>>>(io, cnt, kBN, kGD);
    row_sum_mask_kernel<<<elts(kBN), TPB, 0, stream>>>(io, io_mask, kBN, kGD);
    // attention 1 (over io)
    gemm_fused(io, a1wf_b, a1_bf, fa1, kBN, kAD, kGD, kGD, kAD, kAD, false, stream);
    gemm_split(h1, a1wd_b, part, kB, kAD, kDD, kDD, kAD, stream);
    combine(part, kSPLIT, kB, kAD, kAD, a1_bd, nullptr, hd1, kAD, stream);
    att_score_kernel<<<elts(kB * kNOBJ), TPB, 0, stream>>>(fa1, hd1, a1_wfull, a1_bfull,
                                                           io_mask, sc1, kNOBJ, kAD);
    softmax_rows_kernel<<<elts(kB), TPB, 0, stream>>>(sc1, kB, kNOBJ);
    weighted_sum_kernel<<<elts(kB * kGD), TPB, 0, stream>>>(io, sc1, gvec, kNOBJ, kGD);
    // attention 2 (over image features; fA2 hoisted)
    concat2_kernel<<<elts((size_t)kB * (kDD + kGD)), TPB, 0, stream>>>(h1, gvec, cat_hg, kB, kDD, kGD);
    gemm_split(cat_hg, a2wd_b, part, kB, kAD, kDD + kGD, kDD + kGD, kAD, stream);
    combine(part, kSPLIT, kB, kAD, kAD, a2_bd, nullptr, hd2, kAD, stream);
    att_score_kernel<<<elts(kB * kR), TPB, 0, stream>>>(fA2, hd2, a2_wfull, a2_bfull,
                                                        nullptr, sc2, kR, kAD);
    softmax_rows_kernel<<<elts(kB), TPB, 0, stream>>>(sc2, kB, kR);
    weighted_sum_kernel<<<elts(kB * kFD), TPB, 0, stream>>>(image_features, sc2, vvec, kR, kFD);
    // LSTM2
    build_x2_kernel<<<elts(kB * kX2), TPB, 0, stream>>>(gvec, vvec, h1, x2);
    gemm_split(x2, wih2t, part,  kB, k4DD, kX2, kX2, k4DD, stream);
    gemm_split(h2, whh2t, partB, kB, k4DD, kDD, kDD, k4DD, stream);
    combine(part, 2 * kSPLIT, kB, k4DD, k4DD, bih2, bhh2, g2, k4DD, stream);
    lstm_pointwise_kernel<<<elts(kB * kDD), TPB, 0, stream>>>(g2, h2, c2);
    // outputs: [2, B, T, V]
    gemm_split(h2, wfc_b, part, kB, kV, kDD, kDD, kVpad, stream);
    combine(part, kSPLIT, kB, kV, kVpad, bfc, nullptr, out + (size_t)t * kV, kT * kV, stream);
    gemm_split(h1, wfc1_b, part, kB, kV, kDD, kDD, kVpad, stream);
    combine(part, kSPLIT, kB, kV, kVpad, bfc1, nullptr,
            out + (size_t)kB * kT * kV + (size_t)t * kV, kT * kV, stream);
  }
}
